// TorchHawkesNNCovariates_84378927497619
// MI455X (gfx1250) — matrix operand order repaired and checked by reading, not compiled, not run on hardware
//
#include <hip/hip_runtime.h>
#include <math.h>

#define KLOC 64     // locations
#define NT   8192   // time steps
#define MF   8      // covariate features
#define DD   32     // memory depth
#define FEAT 512    // 2*D*M
#define HID  20     // hidden units
#define HPAD 32     // hidden padded to 2 WMMA column tiles
#define TILE_N 128  // time steps per block (8 waves x 16 rows)
#define WIN (TILE_N + 2 * DD)  // 192 staged covariate rows

typedef __attribute__((ext_vector_type(16))) _Float16 v16h;
typedef __attribute__((ext_vector_type(8)))  _Float16 v8h;
typedef __attribute__((ext_vector_type(4)))  _Float16 v4h;
typedef __attribute__((ext_vector_type(8)))  float    v8f;

__device__ __forceinline__ float softplusf(float x) {
    // numerically stable: max(x,0) + log1p(exp(-|x|))
    return fmaxf(x, 0.0f) + log1pf(expf(-fabsf(x)));
}

// ---------------- Kernel 1: sequential Hawkes excitation scan ----------------
// S_all[t, k] = S before update;  S <- decay * (S + beta * obs[k, t])
// 1 block x 256 threads: obs is staged through LDS in 64x64 tiles so global
// reads are coalesced; lanes 0..63 carry the recurrence (lane = location k).
__global__ void __launch_bounds__(256)
hawkes_scan_kernel(const float* __restrict__ obs,
                   const float* __restrict__ hbeta,
                   float* __restrict__ S_all) {
    __shared__ float sObs[64 * KLOC];     // [t-in-tile][k], 16 KB
    const int tid = threadIdx.x;
    const int k = tid & 63;
    const float beta  = hbeta[0];
    const float decay = expf(-beta);
    float S = 0.0f;

    for (int t0 = 0; t0 < NT; t0 += 64) {
        // coalesced stage: consecutive tid -> consecutive t for one k-row
        for (int i = tid; i < 64 * KLOC; i += 256) {
            const int kk = i >> 6, tt = i & 63;
            sObs[tt * KLOC + kk] = obs[(size_t)kk * NT + t0 + tt];
        }
        __syncthreads();
        if (tid < KLOC) {
            for (int tt = 0; tt < 64; ++tt) {
                S_all[(size_t)(t0 + tt) * KLOC + k] = S;
                S = decay * (S + beta * sObs[tt * KLOC + k]);
            }
        }
        __syncthreads();
    }
}

// ---------------- Kernel 2: lam1 = softplus(halpha @ S_all^T) ----------------
// Written into the lams output slot; kernel 3 adds gamma*mu on top.
__global__ void lam1_kernel(const float* __restrict__ halpha,
                            const float* __restrict__ S_all,
                            float* __restrict__ lams) {
    const int idx = blockIdx.x * 256 + threadIdx.x;   // over K*N
    const int k = idx >> 13;                          // N = 8192 = 2^13
    const int n = idx & (NT - 1);
    const float* __restrict__ hr = halpha + k * KLOC;
    const float* __restrict__ sr = S_all + (size_t)n * KLOC;
    float acc = 0.0f;
#pragma unroll 8
    for (int j = 0; j < KLOC; ++j) acc += hr[j] * sr[j];
    lams[idx] = softplusf(acc);
}

// ---------------- Kernel 3: sliding-window MLP via WMMA ----------------
// grid.x = KLOC * (NT / TILE_N); block = 256 (8 waves, wave32)
__global__ void __launch_bounds__(256)
mlp_wmma_kernel(const float* __restrict__ covs,
                const float* __restrict__ obs,
                const float* __restrict__ gamma,
                const float* __restrict__ W1,
                const float* __restrict__ b1,
                const float* __restrict__ W2,
                const float* __restrict__ b2,
                float* __restrict__ lams,       // [K,N]; holds lam1 on entry
                float* __restrict__ partials)   // [gridDim.x] loglik partials
{
    // W1 padded to 32 rows (rows 20..31 = 0) so the hot loop needs no masking
    __shared__ __align__(32) _Float16 sW1p[HPAD * FEAT];   // 32 KB
    __shared__ __align__(16) _Float16 sCov[WIN * MF];      // 3 KB
    __shared__ float sB1[HID];
    __shared__ float sW2[HID];
    __shared__ float sPart[8];

    const int k  = blockIdx.x >> 6;             // location (64 tiles per k)
    const int n0 = (blockIdx.x & 63) * TILE_N;  // tile start time

    // Stage W1 (f32 -> f16), vectorized: float4 load -> packed v4h store
    {
        const float4* __restrict__ W1v = (const float4*)W1;
        v4h* __restrict__ dst = (v4h*)sW1p;
        for (int i = threadIdx.x; i < (HID * FEAT) / 4; i += 256) {
            const float4 w = W1v[i];
            v4h h;
            h[0] = (_Float16)w.x; h[1] = (_Float16)w.y;
            h[2] = (_Float16)w.z; h[3] = (_Float16)w.w;
            dst[i] = h;
        }
        // zero pad rows 20..31 (dword stores)
        unsigned int* __restrict__ zp = (unsigned int*)(sW1p + HID * FEAT);
        for (int i = threadIdx.x; i < ((HPAD - HID) * FEAT) / 2; i += 256)
            zp[i] = 0u;
    }
    if (threadIdx.x < HID) {
        sB1[threadIdx.x] = b1[threadIdx.x];
        sW2[threadIdx.x] = W2[threadIdx.x];
    }
    // Stage covariate window rows [n0-32, n0+160), edge-clamped; one row = 8
    // floats = 2 float4 loads -> 2 v4h stores.
    {
        const float4* __restrict__ cv =
            (const float4*)(covs + (size_t)k * NT * MF);
        v4h* __restrict__ dst = (v4h*)sCov;
        for (int i = threadIdx.x; i < WIN * 2; i += 256) {
            const int r = i >> 1, hrow = i & 1;
            int g = n0 + r - DD;
            g = (g < 0) ? 0 : ((g > NT - 1) ? NT - 1 : g);
            const float4 w = cv[g * 2 + hrow];
            v4h h;
            h[0] = (_Float16)w.x; h[1] = (_Float16)w.y;
            h[2] = (_Float16)w.z; h[3] = (_Float16)w.w;
            dst[i] = h;
        }
    }
    __syncthreads();

    const int wave  = threadIdx.x >> 5;
    const int lane  = threadIdx.x & 31;
    const int lrow  = lane & 15;   // A row offset / B-C column index
    const int hi    = lane >> 4;   // half-wave select
    const int rbase = wave * 16;   // 16 time rows per wave

    // Contiguous-LDS views:
    //  - one covariate row (8 halfs, 16 B) == one v8h
    //  - one B fragment == 32 contiguous bytes of a W1 row
    const v8h* __restrict__ covRows = (const v8h*)sCov;
    const v8h* __restrict__ w1Row0  = (const v8h*)(sW1p + lrow * FEAT);
    const v8h* __restrict__ w1Row1  = (const v8h*)(sW1p + (16 + lrow) * FEAT);
    const int arow = rbase + lrow;

    v8f c0 = {};   // hidden cols 0..15
    v8f c1 = {};   // hidden cols 16..31 (20..31 multiply by staged zeros)

#pragma unroll
    for (int q = 0; q < 16; ++q) {   // 16 chunks of 32 features
        // A fragment: elems 0..15 = feats q*32+hi*16+(0..15)
        // = covariate rows (arow + 4q + 2hi) and (arow + 4q + 2hi + 1).
        const v8h alo = covRows[arow + 4 * q + 2 * hi];
        const v8h ahi = covRows[arow + 4 * q + 2 * hi + 1];
        // B fragment (same contiguous k order as A): elems 0..15 = feats
        // q*32 + hi*16 + (0..15) of column (lane&15) -> 32 contiguous bytes.
        const v8h blo0 = w1Row0[q * 4 + hi * 2];
        const v8h bhi0 = w1Row0[q * 4 + hi * 2 + 1];
        const v8h blo1 = w1Row1[q * 4 + hi * 2];
        const v8h bhi1 = w1Row1[q * 4 + hi * 2 + 1];

        const v16h a  = __builtin_shufflevector(alo, ahi,
            0,1,2,3,4,5,6,7,8,9,10,11,12,13,14,15);
        const v16h b0 = __builtin_shufflevector(blo0, bhi0,
            0,1,2,3,4,5,6,7,8,9,10,11,12,13,14,15);
        const v16h b1f = __builtin_shufflevector(blo1, bhi1,
            0,1,2,3,4,5,6,7,8,9,10,11,12,13,14,15);

        c0 = __builtin_amdgcn_wmma_f32_16x16x32_f16(false, a, false, b0,
                                                    (short)0, c0, false, false);
        c1 = __builtin_amdgcn_wmma_f32_16x16x32_f16(false, a, false, b1f,
                                                    (short)0, c1, false, false);
    }

    // Epilogue: softplus -> W2 dot (cross-lane reduce) -> softplus -> lams/loglik
    const float gk    = gamma[k];
    const float bias2 = b2[0];
    float ll = 0.0f;

    const float w2a  = sW2[lrow];
    const float b1a  = sB1[lrow];
    const float w2bv = (lrow < HID - 16) ? sW2[16 + lrow] : 0.0f;
    const float b1bv = (lrow < HID - 16) ? sB1[16 + lrow] : 0.0f;

#pragma unroll
    for (int v = 0; v < 8; ++v) {
        // C/D layout: VGPR v holds row v (lanes 0-15) / row v+8 (lanes 16-31),
        // column = lane & 15
        float p = softplusf(c0[v] + b1a) * w2a
                + softplusf(c1[v] + b1bv) * w2bv;  // pad cols contribute 0
#pragma unroll
        for (int msk = 1; msk < 16; msk <<= 1)
            p += __shfl_xor(p, msk, 32);           // reduce 16 cols per group
        if (lrow == 0) {
            const int n = n0 + rbase + v + hi * 8;
            const size_t idx = (size_t)k * NT + n;
            const float mu  = softplusf(p + bias2);
            const float lam = lams[idx] + gk * mu;
            lams[idx] = lam;
            ll += obs[idx] * logf(lam) - lam;
        }
    }

    // Block-level log-likelihood partial
#pragma unroll
    for (int msk = 16; msk >= 1; msk >>= 1) ll += __shfl_xor(ll, msk, 32);
    if (lane == 0) sPart[wave] = ll;
    __syncthreads();
    if (threadIdx.x == 0) {
        float s = 0.0f;
#pragma unroll
        for (int w = 0; w < 8; ++w) s += sPart[w];
        partials[blockIdx.x] = s;
    }
}

// ---------------- Kernel 4: final loglik reduction ----------------
__global__ void loglik_reduce_kernel(const float* __restrict__ partials,
                                     int n, float* __restrict__ out) {
    __shared__ float s[256];
    float acc = 0.0f;
    for (int i = threadIdx.x; i < n; i += 256) acc += partials[i];
    s[threadIdx.x] = acc;
    __syncthreads();
    for (int st = 128; st > 0; st >>= 1) {
        if (threadIdx.x < st) s[threadIdx.x] += s[threadIdx.x + st];
        __syncthreads();
    }
    if (threadIdx.x == 0) out[0] = s[0];
}

extern "C" void kernel_launch(void* const* d_in, const int* in_sizes, int n_in,
                              void* d_out, int out_size, void* d_ws, size_t ws_size,
                              hipStream_t stream) {
    (void)in_sizes; (void)n_in; (void)out_size; (void)ws_size;

    const float* obs    = (const float*)d_in[0];  // [K, N]
    const float* covs   = (const float*)d_in[1];  // [K, N, M]
    const float* hbeta  = (const float*)d_in[2];  // [1]
    const float* halpha = (const float*)d_in[3];  // [K, K]
    const float* gamma  = (const float*)d_in[4];  // [K]
    const float* W1     = (const float*)d_in[5];  // [20, 512]
    const float* b1     = (const float*)d_in[6];  // [20]
    const float* W2     = (const float*)d_in[7];  // [1, 20]
    const float* b2     = (const float*)d_in[8];  // [1]

    float* out    = (float*)d_out;
    float* loglik = out;           // scalar first (return order)
    float* lams   = out + 1;       // [K, N]

    float* S_all    = (float*)d_ws;                   // N*K floats (2 MB)
    float* partials = S_all + (size_t)NT * KLOC;      // 4096 floats

    const int nblocks_mlp = KLOC * (NT / TILE_N);     // 4096

    hawkes_scan_kernel<<<1, 256, 0, stream>>>(obs, hbeta, S_all);
    lam1_kernel<<<(KLOC * NT) / 256, 256, 0, stream>>>(halpha, S_all, lams);
    mlp_wmma_kernel<<<nblocks_mlp, 256, 0, stream>>>(covs, obs, gamma,
                                                     W1, b1, W2, b2,
                                                     lams, partials);
    loglik_reduce_kernel<<<1, 256, 0, stream>>>(partials, nblocks_mlp, loglik);
}
